// Attention_9526237463088
// MI455X (gfx1250) — matrix operand order repaired and checked by reading, not compiled, not run on hardware
//
#include <hip/hip_runtime.h>
#include <hip/hip_bf16.h>

// ---------------------------------------------------------------------------
// CDNA5 (gfx1250) fused multi-head attention, bf16 WMMA path.
//   qp = q @ Wq^T + bq   (f32 -> bf16, ws)
//   kp, vp likewise
//   ctx = flash-attention(qp, kp, vp)   (bf16, ws)
//   out = ctx @ Wo^T + bo (f32, d_out)
// ---------------------------------------------------------------------------

typedef __attribute__((ext_vector_type(16))) __bf16 v16bf;
typedef __attribute__((ext_vector_type(8)))  float  v8f;

#if __has_builtin(__builtin_amdgcn_global_load_async_to_lds_b128) && \
    __has_builtin(__builtin_amdgcn_s_wait_asynccnt)
#define HAVE_ASYNC_LDS 1
typedef __attribute__((__vector_size__(4 * sizeof(int)))) int v4i_t;
typedef __attribute__((address_space(1))) v4i_t* gvec_t;
typedef __attribute__((address_space(3))) v4i_t* lvec_t;
#else
#define HAVE_ASYNC_LDS 0
#endif

union Frag {
    v16bf bf;
    uint4 q[2];
    unsigned int   u[8];
    unsigned short h[16];
};
union H8 {
    uint4 q;
    unsigned short h[8];
};

__device__ __forceinline__ unsigned short f2b(float x) {
    union { float f; unsigned int u; } t; t.f = x;
    unsigned int r = t.u + 0x7FFFu + ((t.u >> 16) & 1u);   // round-to-nearest-even
    return (unsigned short)(r >> 16);
}

// Pack two f32 -> packed bf16 pair (low = x, high = y).
#if __has_builtin(__builtin_amdgcn_cvt_pk_bf16_f32)
__device__ __forceinline__ unsigned int pk_bf16(float x, float y) {
    auto r = __builtin_amdgcn_cvt_pk_bf16_f32(x, y);
    unsigned int u;
    __builtin_memcpy(&u, &r, 4);
    return u;
}
#else
__device__ __forceinline__ unsigned int pk_bf16(float x, float y) {
    return (unsigned int)f2b(x) | ((unsigned int)f2b(y) << 16);
}
#endif

__device__ __forceinline__ v8f wmma_bf16(v16bf a, v16bf b, v8f c) {
    return __builtin_amdgcn_wmma_f32_16x16x32_bf16(
        false, a, false, b, (short)0, c, false, false);
}

// ---- DPP 16-lane reductions (VALU only, no LDS) ---------------------------
template<int CTRL>
__device__ __forceinline__ float dpp_movf(float x) {
    return __int_as_float(__builtin_amdgcn_update_dpp(
        0, __float_as_int(x), CTRL, 0xF, 0xF, true));
}
// quad_perm [1,0,3,2]=0xB1 (xor1), [2,3,0,1]=0x4E (xor2), row_ror:4/8
__device__ __forceinline__ float red16_max(float v) {
    v = fmaxf(v, dpp_movf<0xB1>(v));
    v = fmaxf(v, dpp_movf<0x4E>(v));
    v = fmaxf(v, dpp_movf<0x124>(v));
    v = fmaxf(v, dpp_movf<0x128>(v));
    return v;
}
__device__ __forceinline__ float red16_sum(float v) {
    v += dpp_movf<0xB1>(v);
    v += dpp_movf<0x4E>(v);
    v += dpp_movf<0x124>(v);
    v += dpp_movf<0x128>(v);
    return v;
}

// A fragment: lanes 0-15 row m=lane, K {0..7,16..23}; lanes 16-31 same row,
// K {8..15,24..31}.
__device__ __forceinline__ v16bf a_frag_lds(const char* tile, int row, int pitch,
                                            int lane, int dbyte) {
    const char* p = tile + (size_t)(row + (lane & 15)) * pitch + dbyte
                    + ((lane & 16) ? 16 : 0);
    Frag f;
    f.q[0] = *(const uint4*)(p);
    f.q[1] = *(const uint4*)(p + 32);
    return f.bf;
}

// B fragment: tile stores B columns as rows (row index = n, 32 contiguous K).
// lanes 0-15 col n=lane K {0..7,16..23}; lanes 16-31 col n=lane-16 K {8..15,24..31}.
__device__ __forceinline__ v16bf b_frag_lds(const char* tile, int colbase, int pitch,
                                            int lane, int dbyte) {
    const char* p = tile + (size_t)(colbase + (lane & 15)) * pitch + dbyte
                    + ((lane & 16) ? 16 : 0);
    Frag f;
    f.q[0] = *(const uint4*)(p);
    f.q[1] = *(const uint4*)(p + 32);
    return f.bf;
}

// ---------------------------------------------------------------------------
// GEMM: out[M,N] = A[M,K] @ W[N,K]^T + bias[N]
// 256 thr = 8 waves; block tile 128(M) x 128(N); K step 32.
// Wave grid 4x2, each wave 32x64 (2x4 WMMA tiles, 8 WMMA per K step).
// ---------------------------------------------------------------------------
#define GP 80   // LDS row pitch (bytes) for 32-bf16 rows

template<int A_BF16, int OUT_BF16>
__global__ __launch_bounds__(256) void gemm_kernel(
    const void* __restrict__ Ap, const float* __restrict__ W,
    const float* __restrict__ bias, void* __restrict__ outp,
    int M, int N, int K)
{
    __shared__ __align__(16) char As[128 * GP];
    __shared__ __align__(16) char Bs[128 * GP];

    const int tid  = threadIdx.x;
    const int lane = tid & 31;
    const int wave = tid >> 5;
    const int wr = wave >> 1, wc = wave & 1;
    const int row0 = blockIdx.y * 128;
    const int col0 = blockIdx.x * 128;

    v8f acc[2][4];
    #pragma unroll
    for (int i = 0; i < 2; ++i)
        #pragma unroll
        for (int j = 0; j < 4; ++j) acc[i][j] = {};

    for (int k0 = 0; k0 < K; k0 += 32) {
        // ---- stage A tile (128 x 32) as bf16 ----
        if (A_BF16) {
            const unsigned short* A = (const unsigned short*)Ap;
            #pragma unroll
            for (int i = 0; i < 2; ++i) {
                int c = tid + i * 256;          // 512 chunks of 8 bf16
                int r = c >> 2, cc = c & 3;
#if HAVE_ASYNC_LDS
                __builtin_amdgcn_global_load_async_to_lds_b128(
                    (gvec_t)(A + (size_t)(row0 + r) * K + k0 + cc * 8),
                    (lvec_t)(As + r * GP + cc * 16), 0, 0);
#else
                *(uint4*)(As + r * GP + cc * 16) =
                    *(const uint4*)(A + (size_t)(row0 + r) * K + k0 + cc * 8);
#endif
            }
        } else {
            const float* A = (const float*)Ap;
            #pragma unroll
            for (int i = 0; i < 4; ++i) {
                int c = tid + i * 256;          // 1024 chunks of 4 f32
                int r = c >> 3, cc = c & 7;
                float4 d = *(const float4*)(A + (size_t)(row0 + r) * K + k0 + cc * 4);
                uint2 p;
                p.x = pk_bf16(d.x, d.y);
                p.y = pk_bf16(d.z, d.w);
                *(uint2*)(As + r * GP + cc * 8) = p;
            }
            if (k0 + 32 < K)
                __builtin_prefetch(A + (size_t)(row0 + (tid >> 1)) * K + k0 + 32, 0, 1);
        }
        // ---- stage B tile (128 n-rows x 32 k) as bf16 ----
        #pragma unroll
        for (int i = 0; i < 4; ++i) {
            int c = tid + i * 256;              // 1024 chunks of 4 f32
            int r = c >> 3, cc = c & 7;
            float4 d = *(const float4*)(W + (size_t)(col0 + r) * K + k0 + cc * 4);
            uint2 p;
            p.x = pk_bf16(d.x, d.y);
            p.y = pk_bf16(d.z, d.w);
            *(uint2*)(Bs + r * GP + cc * 8) = p;
        }
        if (k0 + 32 < K)
            __builtin_prefetch(W + (size_t)(col0 + (tid >> 1)) * K + k0 + 32, 0, 1);
#if HAVE_ASYNC_LDS
        if (A_BF16) __builtin_amdgcn_s_wait_asynccnt(0);
#endif
        __syncthreads();

        // load all fragments first, then issue the WMMA chain
        v16bf af[2], bfr[4];
        #pragma unroll
        for (int ms = 0; ms < 2; ++ms)
            af[ms] = a_frag_lds(As, wr * 32 + ms * 16, GP, lane, 0);
        #pragma unroll
        for (int ns = 0; ns < 4; ++ns)
            bfr[ns] = b_frag_lds(Bs, wc * 64 + ns * 16, GP, lane, 0);

        #pragma unroll
        for (int ms = 0; ms < 2; ++ms)
            #pragma unroll
            for (int ns = 0; ns < 4; ++ns)
                acc[ms][ns] = wmma_bf16(af[ms], bfr[ns], acc[ms][ns]);
        __syncthreads();
    }

    // ---- epilogue: C layout lane=n, vgpr i -> m = (lane>>4)*8 + i ----
    #pragma unroll
    for (int ms = 0; ms < 2; ++ms)
        #pragma unroll
        for (int ns = 0; ns < 4; ++ns) {
            int n     = col0 + wc * 64 + ns * 16 + (lane & 15);
            int mbase = row0 + wr * 32 + ms * 16 + ((lane >> 4) << 3);
            float bv = bias[n];
            #pragma unroll
            for (int i = 0; i < 8; ++i) {
                float v = acc[ms][ns][i] + bv;
                if (OUT_BF16)
                    ((unsigned short*)outp)[(size_t)(mbase + i) * N + n] = f2b(v);
                else
                    ((float*)outp)[(size_t)(mbase + i) * N + n] = v;
            }
        }
}

// ---------------------------------------------------------------------------
// Flash attention. Block = 256 thr (8 waves) per (b,h,128 q-rows).
// k-tile = 64 columns: 16 WMMAs per wave per softmax update.
// ---------------------------------------------------------------------------
__global__ __launch_bounds__(256) void attn_kernel(
    const unsigned short* __restrict__ qp, const unsigned short* __restrict__ kp,
    const unsigned short* __restrict__ vp, unsigned short* __restrict__ ctx)
{
    constexpr int L = 2048, D = 1024;
    constexpr int KP = 144;   // pitch for 64-bf16 rows (128B data + 16 pad)

    __shared__ __align__(16) char Kt[64 * KP];       // [seq 0..63][d 0..63]
    __shared__ __align__(16) char Vt[64 * KP];       // transposed: [d 0..63][seq 0..63]
    __shared__ __align__(16) char Pb[8 * 16 * KP];   // per-wave P (16 x 64)

    const int tid = threadIdx.x, lane = tid & 31, wave = tid >> 5;
    const int b = blockIdx.y >> 4, h = blockIdx.y & 15;
    const int q0 = blockIdx.x * 128;
    const size_t base = ((size_t)b * L) * D + (size_t)h * 64;

    // ---- Q fragments (16 rows x 64 d) in registers for whole kernel ----
    const int qrow = q0 + wave * 16 + (lane & 15);
    const unsigned short* qptr = qp + base + (size_t)qrow * D;
    const int he = (lane & 16) ? 8 : 0;
    Frag qa[2];
    #pragma unroll
    for (int dc = 0; dc < 2; ++dc) {
        qa[dc].q[0] = *(const uint4*)(qptr + dc * 32 + he);
        qa[dc].q[1] = *(const uint4*)(qptr + dc * 32 + he + 16);
    }

    float mrow[8], lrow[8], alpha[8];
    v8f accO[4];
    #pragma unroll
    for (int i = 0; i < 8; ++i) { mrow[i] = -1e30f; lrow[i] = 0.0f; }
    #pragma unroll
    for (int d = 0; d < 4; ++d) accO[d] = {};

    char* Pw = Pb + wave * 16 * KP;

    for (int kt0 = 0; kt0 < L; kt0 += 64) {
        // ---- stage K (async global->LDS) and V (transposed via VGPR) tiles ----
        #pragma unroll
        for (int i = 0; i < 2; ++i) {
            int c = tid + i * 256;              // 512 chunks of 8 bf16
            int r = c >> 3, cc = c & 7;
            const size_t g = base + (size_t)(kt0 + r) * D + cc * 8;
#if HAVE_ASYNC_LDS
            __builtin_amdgcn_global_load_async_to_lds_b128(
                (gvec_t)(kp + g), (lvec_t)(Kt + r * KP + cc * 16), 0, 0);
#else
            *(uint4*)(Kt + r * KP + cc * 16) = *(const uint4*)(kp + g);
#endif
            H8 t; t.q = *(const uint4*)(vp + g);
            #pragma unroll
            for (int j = 0; j < 8; ++j)
                *(unsigned short*)(Vt + (size_t)(cc * 8 + j) * KP + r * 2) = t.h[j];
        }
        if (kt0 + 64 < L)
            __builtin_prefetch(kp + base + (size_t)(kt0 + 64 + (tid >> 2)) * D, 0, 1);
#if HAVE_ASYNC_LDS
        __builtin_amdgcn_s_wait_asynccnt(0);
#endif
        __syncthreads();

        // ---- S(16x64) = Q(16x64) @ K_tile^T ----
        v16bf kb[4][2];
        #pragma unroll
        for (int ns = 0; ns < 4; ++ns)
            #pragma unroll
            for (int dc = 0; dc < 2; ++dc)
                kb[ns][dc] = b_frag_lds(Kt, ns * 16, KP, lane, dc * 64);
        v8f s[4];
        #pragma unroll
        for (int ns = 0; ns < 4; ++ns) {
            v8f z = {};
            z = wmma_bf16(qa[0].bf, kb[ns][0], z);
            s[ns] = wmma_bf16(qa[1].bf, kb[ns][1], z);
        }

        // ---- online softmax (DPP reductions; lanes 0-15 m 0..7, 16-31 m 8..15)
        #pragma unroll
        for (int i = 0; i < 8; ++i) {
            float a0 = s[0][i] * 0.125f;        // 1/sqrt(64)
            float a1 = s[1][i] * 0.125f;
            float a2 = s[2][i] * 0.125f;
            float a3 = s[3][i] * 0.125f;
            float vmax = red16_max(fmaxf(fmaxf(a0, a1), fmaxf(a2, a3)));
            float mnew = fmaxf(mrow[i], vmax);
            float al = __expf(mrow[i] - mnew);
            float p0 = __expf(a0 - mnew);
            float p1 = __expf(a1 - mnew);
            float p2 = __expf(a2 - mnew);
            float p3 = __expf(a3 - mnew);
            float rs = red16_sum(p0 + p1 + p2 + p3);
            lrow[i] = lrow[i] * al + rs;
            mrow[i] = mnew;
            alpha[i] = al;
            // write P in A-matrix memory layout (row m, 64 contiguous k)
            int m = ((lane >> 4) << 3) + i;
            int n = lane & 15;
            unsigned short* pr = (unsigned short*)(Pw + m * KP);
            unsigned int u01 = pk_bf16(p0, p1);
            unsigned int u23 = pk_bf16(p2, p3);
            pr[n]      = (unsigned short)u01;
            pr[16 + n] = (unsigned short)(u01 >> 16);
            pr[32 + n] = (unsigned short)u23;
            pr[48 + n] = (unsigned short)(u23 >> 16);
        }

        // ---- rescale running O, then O += P(16x64) @ V(64x64) ----
        #pragma unroll
        for (int ds = 0; ds < 4; ++ds)
            #pragma unroll
            for (int i = 0; i < 8; ++i)
                accO[ds][i] *= alpha[i];

        v16bf pa[2];
        #pragma unroll
        for (int kc = 0; kc < 2; ++kc)
            pa[kc] = a_frag_lds(Pw, 0, KP, lane, kc * 64);
        v16bf vb[4][2];
        #pragma unroll
        for (int ds = 0; ds < 4; ++ds)
            #pragma unroll
            for (int kc = 0; kc < 2; ++kc)
                vb[ds][kc] = b_frag_lds(Vt, ds * 16, KP, lane, kc * 64);
        #pragma unroll
        for (int ds = 0; ds < 4; ++ds) {
            accO[ds] = wmma_bf16(pa[0], vb[ds][0], accO[ds]);
            accO[ds] = wmma_bf16(pa[1], vb[ds][1], accO[ds]);
        }
        __syncthreads();
    }

    // ---- normalize and store bf16 context (heads re-merged) ----
    #pragma unroll
    for (int i = 0; i < 8; ++i) {
        float inv = 1.0f / lrow[i];
        int m = q0 + wave * 16 + ((lane >> 4) << 3) + i;
        unsigned short* orow = ctx + base + (size_t)m * D;
        #pragma unroll
        for (int ds = 0; ds < 4; ++ds)
            orow[ds * 16 + (lane & 15)] = f2b(accO[ds][i] * inv);
    }
}

// ---------------------------------------------------------------------------
extern "C" void kernel_launch(void* const* d_in, const int* in_sizes, int n_in,
                              void* d_out, int out_size, void* d_ws, size_t ws_size,
                              hipStream_t stream) {
    const float* q   = (const float*)d_in[0];
    const float* k   = (const float*)d_in[1];
    const float* v   = (const float*)d_in[2];
    const float* w_q = (const float*)d_in[3];
    const float* b_q = (const float*)d_in[4];
    const float* w_k = (const float*)d_in[5];
    const float* b_k = (const float*)d_in[6];
    const float* w_v = (const float*)d_in[7];
    const float* b_v = (const float*)d_in[8];
    const float* w_o = (const float*)d_in[9];
    const float* b_o = (const float*)d_in[10];

    const int B = 4, L = 2048, D = 1024;
    const int M = B * L;                       // 8192
    const size_t sz = (size_t)M * D;

    unsigned short* qp = (unsigned short*)d_ws;
    unsigned short* kp = qp + sz;
    unsigned short* vp = kp + sz;
    unsigned short* cx = vp + sz;

    dim3 gg(D / 128, M / 128);                 // (8, 64)
    gemm_kernel<0, 1><<<gg, 256, 0, stream>>>(q, w_q, b_q, qp, M, D, D);
    gemm_kernel<0, 1><<<gg, 256, 0, stream>>>(k, w_k, b_k, kp, M, D, D);
    gemm_kernel<0, 1><<<gg, 256, 0, stream>>>(v, w_v, b_v, vp, M, D, D);

    attn_kernel<<<dim3(L / 128, B * 16), 256, 0, stream>>>(qp, kp, vp, cx);

    gemm_kernel<1, 0><<<gg, 256, 0, stream>>>(cx, w_o, b_o, (float*)d_out, M, D, D);
}
